// Kernel_4740052357514212317_17291538333974
// MI455X (gfx1250) — matrix operand order repaired and checked by reading, not compiled, not run on hardware
//
#include <hip/hip_runtime.h>
#include <hip/hip_bf16.h>

// ---------------------------------------------------------------------------
// Problem constants (reference: N,C,H,W = 32,256,56,56)
// ---------------------------------------------------------------------------
#define NB   32
#define CC   256
#define HH   56
#define WW   56
#define WP   64              // padded row width (4 zeros each side)
#define HWP  (HH * WP)       // 3584  (padded K' domain for GEMM1, N' domain for GEMM2)
#define HW   (HH * WW)       // 3136
#define C5   (CC * 5)        // 1280
#define XPELEMS ((long long)NB * CC * HWP)   // 29,360,128 bf16 elements
#define GUARD 128            // guard elements (256 bytes) each side of Xp

typedef __attribute__((ext_vector_type(16))) __bf16 v16bf;
typedef __attribute__((ext_vector_type(8)))  float  v8f;

union BFrag {
    v16bf        v;
    uint4        q[2];
    unsigned int u[8];
    __bf16       h[16];
};

// ---------------------------------------------------------------------------
// Prep: x (f32, NCHW) -> Xp (bf16, rows padded 56 -> 64 with zeros), plus
// zeroed 256B guards on both ends (OOB neighbor reads must multiply as 0, so
// guards must be non-NaN). x is read exactly once -> nontemporal loads.
// ---------------------------------------------------------------------------
__global__ void __launch_bounds__(256) prep_xpad(const float* __restrict__ x,
                                                 __bf16* __restrict__ base) {
    long long tid = (long long)blockIdx.x * blockDim.x + threadIdx.x;
    const long long total = XPELEMS + 2 * GUARD;
    if (tid >= total) return;
    if (tid < GUARD || tid >= XPELEMS + GUARD) { base[tid] = (__bf16)0.0f; return; }
    long long j = tid - GUARD;
    int p = (int)(j & 63);               // padded column
    long long row = j >> 6;              // flat [n][c][h]
    float v = 0.0f;
    if (p >= 4 && p < 60) v = __builtin_nontemporal_load(&x[row * WW + (p - 4)]);
    base[tid] = (__bf16)v;
}

// ---------------------------------------------------------------------------
// GEMM1: t6[n][m][nn] = bf16( (1/56) * p6w[m][nn] * sum_{k'} Xp[m][k'] * Xp[cin][k' + 2*tap - 4] )
// M = 256 (c_out), N = 1280 (cin*5 + tap), K' = 3584 (padded hw).
// Both operands K-contiguous -> direct global fragment loads, no LDS; wave
// redundancy is served by WGP$/L2. WG tile 128x128, wave tile 64x32.
// ---------------------------------------------------------------------------
__global__ void __launch_bounds__(256) gemm1_corr(const __bf16* __restrict__ Xp,
                                                  const float* __restrict__ p6w,
                                                  __bf16* __restrict__ t6) {
    const int n    = blockIdx.y;
    const int tile = blockIdx.x;              // 0..19
    const int m0   = (tile & 1) * 128;
    const int n0   = (tile >> 1) * 128;
    const int tid  = threadIdx.x;
    const int wid  = tid >> 5, lane = tid & 31;
    const int wm   = wid & 1,  wn   = wid >> 1;   // wave grid 2(M) x 4(N)
    const int l16  = lane & 15;
    const int kh   = lane >> 4;                   // lane-half selects K sub-range

    const __bf16* Xb = Xp + (long long)n * CC * HWP;

    // A rows (M = l16 within each 16-row fragment)
    const __bf16* arow[4];
#pragma unroll
    for (int mi = 0; mi < 4; ++mi) {
        int m = m0 + wm * 64 + mi * 16 + l16;
        arow[mi] = Xb + (long long)m * HWP;
    }
    // B columns: nn -> (cin, tap); column is Xp row shifted by 2*tap-4.
    // B layout: N = l16, K = e + 16*kh  -> one contiguous 16-element run.
    const __bf16* bcol[2];
    int nnv[2];
#pragma unroll
    for (int ni = 0; ni < 2; ++ni) {
        int nn = n0 + wn * 32 + ni * 16 + l16;
        nnv[ni] = nn;
        int cin = nn / 5, tap = nn % 5;
        bcol[ni] = Xb + (long long)cin * HWP + (2 * tap - 4) + 16 * kh;
    }

    v8f acc[4][2];
#pragma unroll
    for (int mi = 0; mi < 4; ++mi)
#pragma unroll
        for (int ni = 0; ni < 2; ++ni)
#pragma unroll
            for (int r = 0; r < 8; ++r) acc[mi][ni][r] = 0.0f;

    for (int k0 = 0; k0 < HWP; k0 += 32) {
        BFrag a[4], b[2];
#pragma unroll
        for (int mi = 0; mi < 4; ++mi) {
            // A layout: K = e + 16*kh (contiguous 16-element run, matches B)
            a[mi].q[0] = *(const uint4*)(arow[mi] + k0 + 16 * kh);
            a[mi].q[1] = *(const uint4*)(arow[mi] + k0 + 16 * kh + 8);
        }
#pragma unroll
        for (int ni = 0; ni < 2; ++ni) {
            const unsigned int* p = (const unsigned int*)(bcol[ni] + k0);  // 4B-aligned
#pragma unroll
            for (int u = 0; u < 8; ++u) b[ni].u[u] = p[u];
        }
#pragma unroll
        for (int mi = 0; mi < 4; ++mi)
#pragma unroll
            for (int ni = 0; ni < 2; ++ni)
                acc[mi][ni] = __builtin_amdgcn_wmma_f32_16x16x32_bf16(
                    false, a[mi].v, false, b[ni].v, (short)0, acc[mi][ni], false, false);
    }

    // Fused gate epilogue: t6 = p6w * t2 / sqrt(hw), sqrt(3136)=56. Store bf16.
    const float s = 1.0f / 56.0f;
#pragma unroll
    for (int mi = 0; mi < 4; ++mi)
#pragma unroll
        for (int ni = 0; ni < 2; ++ni) {
            int nn = nnv[ni];
#pragma unroll
            for (int r = 0; r < 8; ++r) {
                int m = m0 + wm * 64 + mi * 16 + r + 8 * kh;   // D: M = r + 8*lanehi
                float g = p6w[m * C5 + nn] * s;
                t6[((long long)n * CC + m) * C5 + nn] = (__bf16)(acc[mi][ni][r] * g);
            }
        }
}

// ---------------------------------------------------------------------------
// GEMM2: out[n][m][h][w] = (1/sqrt(1280)) * sum_nn t6[m][nn] * gelu(relu(x[cin][h+tap-2][w]))
// M = 256, N' = 3584 (padded), K = 1280. A = t6 (K-contiguous, direct loads).
// B staged into a DOUBLE-BUFFERED column-major LDS tile: iteration k stages
// tile k+1 (global loads + gelu VALU + ds stores) while the matrix pipe chews
// tile k -> transcendental staging co-executes with WMMA, one barrier/step.
// Only w<56 columns stored; output is write-once -> nontemporal stores.
// ---------------------------------------------------------------------------
__device__ __forceinline__ float gelu_relu(float x) {
    if (x <= 0.0f) return 0.0f;
    return 0.5f * x * (1.0f + erff(x * 0.70710678118654752f));
}

#define BT_STRIDE 40   // per-column LDS stride (bf16 elems); 80B keeps 16B alignment, skews banks

__global__ void __launch_bounds__(256) gemm2_spread(const __bf16* __restrict__ Xp,
                                                    const __bf16* __restrict__ t6,
                                                    float* __restrict__ out) {
    __shared__ __align__(16) __bf16 Bt[2][128 * BT_STRIDE];   // 2 x 10 KB

    const int n    = blockIdx.y;
    const int tile = blockIdx.x;              // 0..55
    const int m0   = (tile & 1) * 128;
    const int i0   = (tile >> 1) * 128;       // padded spatial base (multiple of 128)
    const int tid  = threadIdx.x;
    const int wid  = tid >> 5, lane = tid & 31;
    const int wm   = wid & 1,  wn   = wid >> 1;
    const int l16  = lane & 15;
    const int kh   = lane >> 4;

    const __bf16* Xb = Xp + (long long)n * CC * HWP;

    // B staging: 32 K-rows x 128 columns = 4096 elems, 16 per thread
    const int nn_l = tid >> 3;                // 0..31 : local K row
    const int ch   = tid & 7;                 // 0..7  : 16-column chunk
    const int colb = ch * 16;
    const int ip   = i0 + colb;               // chunk stays inside one padded row (16 | 64)
    const int sh   = ip >> 6;                 // output h
    const int sw   = ip & 63;                 // output w base

    auto stage = [&](int kk, __bf16* buf) {
        int nn = kk + nn_l;
        int cin = nn / 5, tap = nn % 5;
        int hh = sh + tap - 2;                      // vertical tap shift
        bool valid = (hh >= 0) & (hh < HH);
        const __bf16* src = Xb + ((long long)cin * HH + (valid ? hh : 0)) * WP + (sw + 4);
#pragma unroll
        for (int j = 0; j < 16; ++j) {
            float v = valid ? gelu_relu((float)src[j]) : 0.0f;
            buf[(colb + j) * BT_STRIDE + nn_l] = (__bf16)v;
        }
    };

    const __bf16* arow[4];
#pragma unroll
    for (int mi = 0; mi < 4; ++mi) {
        int m = m0 + wm * 64 + mi * 16 + l16;
        arow[mi] = t6 + ((long long)n * CC + m) * C5;
    }

    v8f acc[4][2];
#pragma unroll
    for (int mi = 0; mi < 4; ++mi)
#pragma unroll
        for (int ni = 0; ni < 2; ++ni)
#pragma unroll
            for (int r = 0; r < 8; ++r) acc[mi][ni][r] = 0.0f;

    stage(0, Bt[0]);
    __syncthreads();

    for (int k0 = 0; k0 < C5; k0 += 32) {
        const int cur = (k0 >> 5) & 1;

        // Prefetch/stage next tile into the other buffer while computing this one.
        if (k0 + 32 < C5) stage(k0 + 32, Bt[cur ^ 1]);

        BFrag a[4], b[2];
#pragma unroll
        for (int mi = 0; mi < 4; ++mi) {
            a[mi].q[0] = *(const uint4*)(arow[mi] + k0 + 16 * kh);
            a[mi].q[1] = *(const uint4*)(arow[mi] + k0 + 16 * kh + 8);
        }
#pragma unroll
        for (int ni = 0; ni < 2; ++ni) {
            int col = wn * 32 + ni * 16 + l16;
            const uint4* p = (const uint4*)(&Bt[cur][col * BT_STRIDE + 16 * kh]); // 16B aligned
            b[ni].q[0] = p[0];
            b[ni].q[1] = p[1];
        }
#pragma unroll
        for (int mi = 0; mi < 4; ++mi)
#pragma unroll
            for (int ni = 0; ni < 2; ++ni)
                acc[mi][ni] = __builtin_amdgcn_wmma_f32_16x16x32_bf16(
                    false, a[mi].v, false, b[ni].v, (short)0, acc[mi][ni], false, false);

        // One barrier per K-step: makes next tile visible, and fences this
        // buffer's reads before it gets overwritten in the next step.
        __syncthreads();
    }

    const float s = 0.02795084971874737f;   // 1/sqrt(1280)
#pragma unroll
    for (int mi = 0; mi < 4; ++mi)
#pragma unroll
        for (int ni = 0; ni < 2; ++ni) {
            int ip2 = i0 + wn * 32 + ni * 16 + l16;
            int w = ip2 & 63;
            if (w < WW) {
                int h = ip2 >> 6;
#pragma unroll
                for (int r = 0; r < 8; ++r) {
                    int m = m0 + wm * 64 + mi * 16 + r + 8 * kh;
                    __builtin_nontemporal_store(acc[mi][ni][r] * s,
                        &out[((long long)n * CC + m) * HW + h * WW + w]);
                }
            }
        }
}

// ---------------------------------------------------------------------------
// Launch
// ---------------------------------------------------------------------------
extern "C" void kernel_launch(void* const* d_in, const int* in_sizes, int n_in,
                              void* d_out, int out_size, void* d_ws, size_t ws_size,
                              hipStream_t stream) {
    (void)in_sizes; (void)n_in; (void)out_size; (void)ws_size;
    const float* x   = (const float*)d_in[0];
    const float* p6w = (const float*)d_in[1];
    float* out = (float*)d_out;

    char* ws = (char*)d_ws;
    __bf16* base = (__bf16*)ws;                 // [guard | Xp | guard]
    __bf16* Xp   = base + GUARD;
    __bf16* t6   = (__bf16*)(ws + (size_t)(XPELEMS + 2 * GUARD) * 2);  // 16B-aligned

    const long long prep_total = XPELEMS + 2 * GUARD;
    const int prep_blocks = (int)((prep_total + 255) / 256);
    prep_xpad<<<prep_blocks, 256, 0, stream>>>(x, base);

    // GEMM1: 2 M-tiles x 10 N-tiles per batch
    gemm1_corr<<<dim3(20, NB), 256, 0, stream>>>(Xp, p6w, t6);

    // GEMM2: 2 M-tiles x 28 N'-tiles per batch
    gemm2_spread<<<dim3(56, NB), 256, 0, stream>>>(Xp, t6, out);
}
